// MultiHeadAttention_54279796687214
// MI455X (gfx1250) — matrix operand order repaired and checked by reading, not compiled, not run on hardware
//
#include <hip/hip_runtime.h>
#include <hip/hip_bf16.h>

typedef __attribute__((ext_vector_type(16))) _Float16 v16h;
typedef __attribute__((ext_vector_type(8)))  _Float16 v8h;
typedef __attribute__((ext_vector_type(8)))  float    v8f;

#define Bsz 64
#define Tsz 256
#define Csz 384
#define Hn  6
#define HSz 64
#define CAT (Hn*HSz)        // 384
#define WE  (Hn*Csz*HSz)    // 147456 elements per weight tensor (Wp same)
#define XS  (Csz + 8)       // LDS row stride (halfs): 784B = 49*16, conflict-free
#define PS  40              // P-staging row stride (halfs): 80B, 16B-aligned, conflict-free

union U16x16 { v16h v; uint4 q[2]; };
union U16x8  { v8h  v; uint4 q;    };

__device__ __forceinline__ v8f wmma_f16(v16h a, v16h b, v8f c) {
    return __builtin_amdgcn_wmma_f32_16x16x32_f16(
        false, a, false, b, (short)0, c, false, false);
}

// Fragment halfs are ascending in K:
//  A (16x32): halfs 0..15 = K 16*half+0..15 (M = l16)
//  B (32x16): halfs 0..15 = K 16*half+0..15 (N = l16)
// -> both are two 16-byte chunks when the K dimension is memory-contiguous.
template <typename P>
__device__ __forceinline__ v16h ld_frag2(const P* p0, const P* p1) {
    U16x16 u;
    u.q[0] = *(const uint4*)p0;
    u.q[1] = *(const uint4*)p1;
    return u.v;
}

__device__ __forceinline__ float red_max16(float v) {
#pragma unroll
    for (int d = 1; d < 16; d <<= 1) v = fmaxf(v, __shfl_xor(v, d, 32));
    return v;
}
__device__ __forceinline__ float red_sum16(float v) {
#pragma unroll
    for (int d = 1; d < 16; d <<= 1) v += __shfl_xor(v, d, 32);
    return v;
}
__device__ __forceinline__ v8f vzero() {
    return (v8f){0.f,0.f,0.f,0.f,0.f,0.f,0.f,0.f};
}

// ---------------------------------------------------------------------------
// Kernel 0: one-shot f32 -> f16 weight conversion with TRANSPOSE.
//   segs 0..2 (Wq,Wk,Wv): [H,C,HS] -> [H,HS,C]   (wT[h][d][c])
//   seg  3    (Wp)      : [CAT,C]  -> [C,CAT]    (wpT[c][k])
// ---------------------------------------------------------------------------
__global__ __launch_bounds__(256)
void wcvt_kernel(const float* __restrict__ Wq, const float* __restrict__ Wk,
                 const float* __restrict__ Wv, const float* __restrict__ Wp,
                 _Float16* __restrict__ w16)
{
    int i = blockIdx.x * 256 + threadIdx.x;
    if (i >= 4 * WE) return;
    int seg = i / WE, off = i % WE;
    size_t dst;
    const float* src;
    if (seg < 3) {
        src = (seg == 0) ? Wq : ((seg == 1) ? Wk : Wv);
        int h = off / (Csz * HSz), rem = off % (Csz * HSz);
        int c = rem / HSz, d = rem % HSz;
        dst = (size_t)seg * WE + (size_t)h * HSz * Csz + (size_t)d * Csz + c;
    } else {
        src = Wp;
        int kk = off / Csz, cc = off % Csz;
        dst = (size_t)3 * WE + (size_t)cc * CAT + kk;
    }
    w16[dst] = (_Float16)src[off];
}

// ---------------------------------------------------------------------------
// Kernel 1: QKV projection.  One block per 16-token tile; x staged once into
// LDS (f16); 8 waves cover the 72 column tiles (9 each).  q,k stored
// [B,H,T,HS]; v stored TRANSPOSED [B,H,HS,T] with packed 16B stores.
// ---------------------------------------------------------------------------
__global__ __launch_bounds__(256)
void qkv_kernel(const float* __restrict__ x,
                const _Float16* __restrict__ w16,   // transposed weights
                _Float16* __restrict__ qo,
                _Float16* __restrict__ ko,
                _Float16* __restrict__ vt)
{
    __shared__ _Float16 xs[16 * XS];
    const int tid = threadIdx.x;
    const int g0  = blockIdx.x * 16;

    for (int p = tid; p < 16 * (Csz / 2); p += 256) {
        int t = p / (Csz / 2), c = 2 * (p % (Csz / 2));
        float2 xv = *(const float2*)(x + (size_t)(g0 + t) * Csz + c);
        union { _Float16 h[2]; unsigned u; } pk;
        pk.h[0] = (_Float16)xv.x; pk.h[1] = (_Float16)xv.y;
        *(unsigned*)&xs[t * XS + c] = pk.u;
    }
    __syncthreads();

    const int lane = tid & 31, half = lane >> 4, l16 = lane & 15;
    const int w = tid >> 5;

    // wave's 9 column tiles: ct = 9w+u ; mat=ct/24, h=(ct%24)/4, n=ct%4
    const _Float16* wp[9];
#pragma unroll
    for (int u = 0; u < 9; ++u) {
        int ct = w * 9 + u;
        int mat = ct / 24, rem = ct % 24, h = rem / 4, n = rem % 4;
        // wT row for output column d = n*16+l16 of head h
        wp[u] = w16 + (size_t)mat * WE + ((size_t)h * HSz + n * 16 + l16) * Csz;
    }

    v8f acc[9];
#pragma unroll
    for (int u = 0; u < 9; ++u) acc[u] = vzero();

    const _Float16* arow = &xs[l16 * XS];
    for (int kb = 0; kb < Csz; kb += 32) {
        v16h a = ld_frag2(arow + kb + 16 * half, arow + kb + 16 * half + 8);
#pragma unroll
        for (int u = 0; u < 9; ++u) {
            const _Float16* bp_ = wp[u] + kb + 16 * half;
            v16h bf = ld_frag2(bp_, bp_ + 8);
            acc[u] = wmma_f16(a, bf, acc[u]);
        }
    }

    const int b  = g0 / Tsz;
    const int t0 = g0 % Tsz;
    const float qscale = 0.05103103630798287f;  // 384^-0.5
#pragma unroll
    for (int u = 0; u < 9; ++u) {
        int ct = w * 9 + u;
        int mat = ct / 24, rem = ct % 24, h = rem / 4, n = rem % 4;
        if (mat == 2) {
            // v transposed: vt[((b*Hn+h)*HSz + d)*Tsz + token], 8 tokens packed
            U16x8 pk;
#pragma unroll
            for (int r = 0; r < 8; ++r) pk.v[r] = (_Float16)acc[u][r];
            _Float16* vb = vt + ((size_t)(b * Hn + h) * HSz + n * 16 + l16) * Tsz
                              + t0 + 8 * half;
            *(uint4*)vb = pk.q;
        } else {
            _Float16* out = (mat == 0) ? qo : ko;
            const float scl = (mat == 0) ? qscale : 1.0f;
            _Float16* ob = out + ((size_t)(b * Hn + h) * Tsz + t0) * HSz
                               + n * 16 + l16;
#pragma unroll
            for (int r = 0; r < 8; ++r) {
                const int row = r + 8 * half;
                ob[(size_t)row * HSz] = (_Float16)(acc[u][r] * scl);
            }
        }
    }
}

// ---------------------------------------------------------------------------
// Kernel 2: flash-style causal attention.  4 waves/block; wave owns one
// 16-row tile of one (b,h).  K is [T,HS] (K-dim = d contiguous); V is
// transposed [HS,T] (K-dim = token contiguous) -> all fragments are 2x b128.
// ---------------------------------------------------------------------------
__global__ __launch_bounds__(128)
void attn_kernel(const _Float16* __restrict__ q,
                 const _Float16* __restrict__ k,
                 const _Float16* __restrict__ vt,
                 _Float16* __restrict__ catted)
{
    __shared__ _Float16 psh[4][16 * PS];
    const int lane = threadIdx.x & 31;
    const int half = lane >> 4;
    const int l16  = lane & 15;
    const int w    = threadIdx.x >> 5;
    const int bh   = blockIdx.x;
    const int b    = bh / Hn;
    const int h    = bh % Hn;
    const int r0   = (blockIdx.y * 4 + w) * 16;

    const _Float16* qp = q  + (size_t)bh * Tsz * HSz;
    const _Float16* kp = k  + (size_t)bh * Tsz * HSz;
    const _Float16* vp = vt + (size_t)bh * HSz * Tsz;   // [HS][T]
    _Float16* psw = &psh[w][0];

    v16h qa[2];
#pragma unroll
    for (int c = 0; c < 2; ++c) {
        const _Float16* base = qp + (size_t)(r0 + l16) * HSz + c * 32;
        qa[c] = ld_frag2(base + 16 * half, base + 16 * half + 8);
    }

    float m_r[8], l_r[8];
#pragma unroll
    for (int r = 0; r < 8; ++r) { m_r[r] = -3.0e38f; l_r[r] = 0.f; }
    v8f o[4];
#pragma unroll
    for (int n = 0; n < 4; ++n) o[n] = vzero();

    const int jbmax = (r0 + 15) / 32;
    for (int jb = 0; jb <= jbmax; ++jb) {
        const int c0 = jb * 32;

        if (jb < jbmax) {   // prefetch next K/V column block
            __builtin_prefetch(kp + (size_t)(c0 + 32 + lane) * HSz, 0, 0);
            __builtin_prefetch(vp + (size_t)lane * Tsz + c0 + 32, 0, 0);
            __builtin_prefetch(vp + (size_t)(lane + 32) * Tsz + c0 + 32, 0, 0);
        }

        // S = Q K^T
        v8f s[2];
#pragma unroll
        for (int sub = 0; sub < 2; ++sub) {
            v8f sa = vzero();
#pragma unroll
            for (int c = 0; c < 2; ++c) {
                const _Float16* base =
                    kp + (size_t)(c0 + sub * 16 + l16) * HSz + c * 32 + 16 * half;
                v16h bf = ld_frag2(base, base + 8);
                sa = wmma_f16(qa[c], bf, sa);
            }
            s[sub] = sa;
        }

        // online softmax (row spans a 16-lane half)
        float alpha[8];
#pragma unroll
        for (int r = 0; r < 8; ++r) {
            const int row  = r0 + r + 8 * half;
            const int col0 = c0 + l16;
            const int col1 = c0 + 16 + l16;
            float e0 = (col0 <= row) ? s[0][r] : -3.0e38f;
            float e1 = (col1 <= row) ? s[1][r] : -3.0e38f;
            float mx = red_max16(fmaxf(e0, e1));
            float mnew = fmaxf(m_r[r], mx);
            float a  = __expf(m_r[r] - mnew);
            float p0 = __expf(e0 - mnew);
            float p1 = __expf(e1 - mnew);
            float rs = red_sum16(p0 + p1);
            l_r[r] = l_r[r] * a + rs;
            m_r[r] = mnew;
            alpha[r] = a;
            const int mrow = r + 8 * half;
            psw[mrow * PS      + l16] = (_Float16)p0;
            psw[mrow * PS + 16 + l16] = (_Float16)p1;
        }
#pragma unroll
        for (int n = 0; n < 4; ++n)
#pragma unroll
            for (int r = 0; r < 8; ++r) o[n][r] *= alpha[r];

        asm volatile("s_wait_dscnt 0" ::: "memory");

        // P as A-fragment from LDS (2x b128)
        const _Float16* prow = &psw[l16 * PS];
        v16h pa = ld_frag2(prow + 16 * half, prow + 16 * half + 8);

        // O += P V  (V^T: K-dim = token contiguous)
#pragma unroll
        for (int n = 0; n < 4; ++n) {
            const _Float16* base =
                vp + (size_t)(n * 16 + l16) * Tsz + c0 + 16 * half;
            v16h bf = ld_frag2(base, base + 8);
            o[n] = wmma_f16(pa, bf, o[n]);
        }
    }

#pragma unroll
    for (int r = 0; r < 8; ++r) l_r[r] = 1.0f / l_r[r];
#pragma unroll
    for (int n = 0; n < 4; ++n)
#pragma unroll
        for (int r = 0; r < 8; ++r) {
            const int row = r0 + r + 8 * half;
            catted[(size_t)(b * Tsz + row) * CAT + h * HSz + n * 16 + l16] =
                (_Float16)(o[n][r] * l_r[r]);
        }
}

// ---------------------------------------------------------------------------
// Kernel 3: output projection with transposed Wp ([C,CAT]).
// ---------------------------------------------------------------------------
__global__ __launch_bounds__(256)
void proj_kernel(const _Float16* __restrict__ catted,
                 const _Float16* __restrict__ wpT,   // [C][CAT]
                 const float* __restrict__ bp,
                 float* __restrict__ out)
{
    __shared__ _Float16 cs[16 * XS];
    const int tid = threadIdx.x;
    const int g0  = blockIdx.x * 16;

    for (int p = tid; p < 16 * (CAT / 2); p += 256) {
        int t = p / (CAT / 2), c = 2 * (p % (CAT / 2));
        *(unsigned*)&cs[t * XS + c] =
            *(const unsigned*)(catted + (size_t)(g0 + t) * CAT + c);
    }
    __syncthreads();

    const int lane = tid & 31, half = lane >> 4, l16 = lane & 15;
    const int w = tid >> 5;

    v8f acc[3];
#pragma unroll
    for (int u = 0; u < 3; ++u) acc[u] = vzero();

    const _Float16* arow = &cs[l16 * XS];
    for (int kb = 0; kb < CAT; kb += 32) {
        v16h a = ld_frag2(arow + kb + 16 * half, arow + kb + 16 * half + 8);
#pragma unroll
        for (int u = 0; u < 3; ++u) {
            const int col = (w * 3 + u) * 16 + l16;
            const _Float16* base = wpT + (size_t)col * CAT + kb + 16 * half;
            v16h bf = ld_frag2(base, base + 8);
            acc[u] = wmma_f16(a, bf, acc[u]);
        }
    }

#pragma unroll
    for (int u = 0; u < 3; ++u) {
        const int col = (w * 3 + u) * 16 + l16;
        const float bias = bp[col];
#pragma unroll
        for (int r = 0; r < 8; ++r) {
            const int row = r + 8 * half;
            out[(size_t)(g0 + row) * Csz + col] = acc[u][r] + bias;
        }
    }
}

// ---------------------------------------------------------------------------
extern "C" void kernel_launch(void* const* d_in, const int* in_sizes, int n_in,
                              void* d_out, int out_size, void* d_ws, size_t ws_size,
                              hipStream_t stream) {
    const float* x  = (const float*)d_in[0];
    const float* Wq = (const float*)d_in[1];
    const float* Wk = (const float*)d_in[2];
    const float* Wv = (const float*)d_in[3];
    const float* Wp = (const float*)d_in[4];
    const float* bp = (const float*)d_in[5];

    const size_t elems = (size_t)Bsz * Hn * Tsz * HSz;   // 6,291,456
    _Float16* qws = (_Float16*)d_ws;
    _Float16* kws = qws + elems;
    _Float16* vws = kws + elems;          // [B,H,HS,T] (transposed)
    _Float16* cat = vws + elems;          // [B,T,H*HS]
    _Float16* w16 = cat + elems;          // [WqT|WkT|WvT|WpT], 4*WE halfs

    wcvt_kernel<<<dim3((4 * WE + 255) / 256), 256, 0, stream>>>(
        Wq, Wk, Wv, Wp, w16);
    qkv_kernel<<<dim3(Bsz * Tsz / 16), 256, 0, stream>>>(
        x, w16, qws, kws, vws);
    attn_kernel<<<dim3(Bsz * Hn, Tsz / 64), 128, 0, stream>>>(
        qws, kws, vws, cat);
    proj_kernel<<<dim3(Bsz * Tsz / 16), 256, 0, stream>>>(
        cat, w16 + (size_t)3 * WE, bp, (float*)d_out);
}
